// CausalSelfAttention_45268955300059
// MI455X (gfx1250) — compile-verified, matrix operand order checked by reading, not run on hardware
//
#include <hip/hip_runtime.h>
#include <hip/hip_bf16.h>

typedef __bf16 bf16;
typedef __attribute__((ext_vector_type(16))) __bf16 v16bf;
typedef __attribute__((ext_vector_type(8)))  float  v8f;
typedef __attribute__((ext_vector_type(4)))  unsigned int v4u;
typedef __attribute__((ext_vector_type(8)))  int    v8i;
typedef __attribute__((ext_vector_type(4)))  int    v4i;

#define SEQ 2048
#define NH  16
#define HD  64
#define DMODEL 1024

// ---------------------------------------------------------------------------
// WMMA helpers
// ---------------------------------------------------------------------------
__device__ __forceinline__ v8f wmma_bf16(v16bf a, v16bf b, v8f c) {
  return __builtin_amdgcn_wmma_f32_16x16x32_bf16(false, a, false, b, (short)0, c,
                                                 false, false);
}

// Load a 16x32 bf16 fragment (A-operand layout; identical pattern serves the
// B-operand when the K x N matrix is stored as N x K row-major).
// ISA layout: lane L (r=L&15, kg=L>>4) holds K = kg*8 + {0..7} in w[0..3]
// and K = 16 + kg*8 + {0..7} in w[4..7]  -> two contiguous uint4 loads.
__device__ __forceinline__ v16bf load_frag(const bf16* __restrict__ tile, int ld) {
  const int lane = threadIdx.x & 31;
  const int r = lane & 15, kg = lane >> 4;
  const uint4* row = (const uint4*)(tile + r * ld);
  union { v16bf v; uint4 q[2]; } f;
  f.q[0] = row[kg];       // bf16 cols kg*8 .. kg*8+7
  f.q[1] = row[2 + kg];   // bf16 cols 16+kg*8 .. 16+kg*8+7
  return f.v;
}

__device__ __forceinline__ unsigned int pack2(float a, float b) {
  union { bf16 h[2]; unsigned int u; } p;
  p.h[0] = (bf16)a; p.h[1] = (bf16)b;
  return p.u;
}

// ---------------------------------------------------------------------------
// Tensor Data Mover: DMA one 128-row x 64-byte tile (32 bf16 per row) from a
// row-major global matrix into LDS, padding each 64B row to 80B (LDS_LD=40)
// so WMMA fragment reads are bank-conflict free.
// D# per CDNA5 ISA 8.3/8.4:
//   group0: count=1 | lds_addr | global_addr(57b) | type=2
//   group1: data_size=3 (8B units), pad_enable, pad_interval=3 (16 DWORDs),
//           pad_amount=3 (4 DWORDs); tensor_dim0=stride, tensor_dim1=128;
//           tile_dim0=8 (64B), tile_dim1=128; tensor_dim0_stride=stride.
// ---------------------------------------------------------------------------
__device__ __forceinline__ void tdm_tile_load(const void* gsrc, unsigned int lds_addr,
                                              unsigned int stride_units) {
  unsigned long long ga = (unsigned long long)(uintptr_t)gsrc;
  v4u g0;
  g0[0] = 1u;                                           // count=1, user mode
  g0[1] = lds_addr;                                     // LDS byte address
  g0[2] = (unsigned int)ga;                             // global_addr[31:0]
  g0[3] = (unsigned int)((ga >> 32) & 0x01FFFFFFu)      // global_addr[56:32]
          | (2u << 30);                                 // type=2 ("image")
  v8i g1;
  g1[0] = (int)((3u << 16) | (1u << 20) | (3u << 22) | (3u << 25));
  g1[1] = (int)((stride_units & 0xFFFFu) << 16);        // tensor_dim0[15:0]
  g1[2] = (int)((stride_units >> 16) | (128u << 16));   // dim0 hi | tensor_dim1 lo
  g1[3] = (int)(8u << 16);                              // tensor_dim1 hi | tile_dim0=8
  g1[4] = 128;                                          // tile_dim1=128, tile_dim2=0
  g1[5] = (int)stride_units;                            // tensor_dim0_stride lo32
  g1[6] = 0;                                            // stride0 hi | stride1 lo
  g1[7] = 0;                                            // stride1 hi
  v4i gz = {0, 0, 0, 0};
#if __clang_major__ >= 23
  v8i gz8 = {0, 0, 0, 0, 0, 0, 0, 0};
  __builtin_amdgcn_tensor_load_to_lds(g0, g1, gz, gz, gz8, 0);
#else
  __builtin_amdgcn_tensor_load_to_lds(g0, g1, gz, gz, 0);
#endif
}

// ---------------------------------------------------------------------------
// Prep kernels
// ---------------------------------------------------------------------------
__global__ void prep_w_kernel(const float* __restrict__ cq,
                              const float* __restrict__ ck,
                              const float* __restrict__ cv,
                              const float* __restrict__ qkvs,
                              bf16* __restrict__ W) {
  int i = blockIdx.x * 256 + threadIdx.x;           // 3072*1024 elements
  int e = i >> 10, d = i & 1023;
  int g = e >> 10, row = e & 1023;
  const float* src = (g == 0) ? cq : ((g == 1) ? ck : cv);
  W[i] = (bf16)(qkvs[e] * src[row * 1024 + d]);
}

__global__ void cvt_kernel(const float* __restrict__ in, bf16* __restrict__ out) {
  int i = blockIdx.x * 256 + threadIdx.x;
  out[i] = (bf16)in[i];
}

__global__ void rot_kernel(float* __restrict__ cosT, float* __restrict__ sinT) {
  int i = blockIdx.x * 256 + threadIdx.x;           // SEQ*32
  int t = i >> 5, j = i & 31;
  float freq = (j < 16) ? powf(1.0f / 1024.0f, (float)j * (1.0f / 15.0f)) : 0.0f;
  float th = (float)t * freq;
  cosT[i] = cosf(th);
  sinT[i] = sinf(th);
}

// ---------------------------------------------------------------------------
// Tiled WMMA GEMM: C[M,N] = A[M,K] * B[N,K]^T   (A,B bf16 row-major, C f32)
// Block = 256 threads (8 waves), tile 128x128, K-step 32.
// Tiles are DMA'd into LDS by the Tensor Data Mover (wave 0 issues, TENSORcnt
// tracks), double-buffered so the next DMA overlaps the current 8 WMMAs.
// TDM pads each 64B row to 80B -> LDS_LD = 40 bf16, conflict-free frag reads.
// ---------------------------------------------------------------------------
#define GBM 128
#define GBN 128
#define GBK 32
#define LDS_LD 40
#define TILE_E   5120     // 128*40 bf16 elements per tile
#define TILE_B  10240     // bytes per tile
#define PAIR_E  10240     // elements per (A,B) buffer pair
#define PAIR_B  20480     // bytes per (A,B) buffer pair

__global__ __launch_bounds__(256)
void gemm_bf16_kernel(const bf16* __restrict__ A, const bf16* __restrict__ B,
                      float* __restrict__ C, int M, int N, int K) {
  extern __shared__ __align__(16) char smem_raw[];   // 2 * PAIR_B = 40960 bytes
  bf16* smem = (bf16*)smem_raw;

  const int tid = threadIdx.x;
  const int wave = tid >> 5, lane = tid & 31;
  const int r = lane & 15, kg = lane >> 4;
  const int mbase = blockIdx.y * GBM, nbase = blockIdx.x * GBN;
  const unsigned int lds_base = (unsigned int)(uintptr_t)smem_raw;
  const unsigned int ku = (unsigned int)(K >> 2);    // row stride in 8B units

  v8f acc[8];
#pragma unroll
  for (int i = 0; i < 8; i++) acc[i] = (v8f){0.f,0.f,0.f,0.f,0.f,0.f,0.f,0.f};

  const bf16* Atile = A + (size_t)mbase * K;
  const bf16* Btile = B + (size_t)nbase * K;
  const int niter = K / GBK;

  if (wave == 0) {                                   // prologue DMA -> buffer 0
    tdm_tile_load(Atile, lds_base, ku);
    tdm_tile_load(Btile, lds_base + TILE_B, ku);
  }

  for (int i = 0; i < niter; i++) {
    const int cur = i & 1;
    if (wave == 0) {
      if (i + 1 < niter) {                           // DMA next pair, then wait
        const unsigned int nb = lds_base + (cur ^ 1) * PAIR_B;
        tdm_tile_load(Atile + (size_t)(i + 1) * GBK, nb, ku);
        tdm_tile_load(Btile + (size_t)(i + 1) * GBK, nb + TILE_B, ku);
        __builtin_amdgcn_s_wait_tensorcnt((short)2); // current pair complete
      } else {
        __builtin_amdgcn_s_wait_tensorcnt((short)0);
      }
    }
    __syncthreads();

    const bf16* As = smem + cur * PAIR_E;
    const bf16* Bs = As + TILE_E;
    v16bf af = load_frag(As + wave * 16 * LDS_LD, LDS_LD);
#pragma unroll
    for (int nt = 0; nt < 8; nt++) {
      v16bf bfrag = load_frag(Bs + nt * 16 * LDS_LD, LDS_LD);
      acc[nt] = wmma_bf16(af, bfrag, acc[nt]);
    }
    __syncthreads();
  }

  // epilogue: C layout lane = n, vgpr rr -> row rr + 8*kg
#pragma unroll
  for (int nt = 0; nt < 8; nt++) {
#pragma unroll
    for (int rr = 0; rr < 8; rr++) {
      int m = mbase + wave * 16 + rr + 8 * kg;
      int n = nbase + nt * 16 + r;
      C[(size_t)m * N + n] = acc[nt][rr];
    }
  }
}

// ---------------------------------------------------------------------------
// Post-QKV: v += softplus(vl)*ve ; RMS-norm q,k ; rotary ; write bf16 layouts
// one wave per (t, head); lane handles dims (lane, lane+32)
// ---------------------------------------------------------------------------
__global__ __launch_bounds__(32)
void post_kernel(const float* __restrict__ qkv, const float* __restrict__ ve,
                 const float* __restrict__ qs, const float* __restrict__ ks,
                 const float* __restrict__ vl,
                 const float* __restrict__ cosT, const float* __restrict__ sinT,
                 bf16* __restrict__ Qb, bf16* __restrict__ Kb,
                 bf16* __restrict__ Vt) {
  const int b = blockIdx.x;
  const int t = b >> 4, h = b & 15;
  const int lane = threadIdx.x;
  const float eps = 1.1920929e-07f;
  const size_t base = (size_t)t * 3072 + h * 64;

  float q0 = qkv[base + lane],        q1 = qkv[base + 32 + lane];
  float k0 = qkv[base + 1024 + lane], k1 = qkv[base + 1024 + 32 + lane];
  float v0 = qkv[base + 2048 + lane], v1 = qkv[base + 2048 + 32 + lane];

  float spl = logf(1.0f + __expf(vl[0]));
  v0 += spl * ve[(size_t)t * 1024 + h * 64 + lane];
  v1 += spl * ve[(size_t)t * 1024 + h * 64 + 32 + lane];
  Vt[(size_t)(h * 64 + lane) * SEQ + t]      = (bf16)v0;   // V^T: [H*64][T]
  Vt[(size_t)(h * 64 + 32 + lane) * SEQ + t] = (bf16)v1;

  float c = cosT[t * 32 + lane], s = sinT[t * 32 + lane];

  // q: RMS-norm (norm over 64 dims) then rotary
  float ssq = q0 * q0 + q1 * q1;
#pragma unroll
  for (int o = 16; o >= 1; o >>= 1) ssq += __shfl_xor(ssq, o, 32);
  float fq = logf(1.0f + __expf(qs[0])) * 8.0f / (sqrtf(ssq) + eps);
  q0 *= fq; q1 *= fq;
  float nq0 =  q0 * c + q1 * s;
  float nq1 = -q0 * s + q1 * c;
  Qb[((size_t)h * SEQ + t) * 64 + lane]      = (bf16)nq0;
  Qb[((size_t)h * SEQ + t) * 64 + 32 + lane] = (bf16)nq1;

  // k
  float ssk = k0 * k0 + k1 * k1;
#pragma unroll
  for (int o = 16; o >= 1; o >>= 1) ssk += __shfl_xor(ssk, o, 32);
  float fk = logf(1.0f + __expf(ks[0])) * 8.0f / (sqrtf(ssk) + eps);
  k0 *= fk; k1 *= fk;
  float nk0 =  k0 * c + k1 * s;
  float nk1 = -k0 * s + k1 * c;
  Kb[((size_t)h * SEQ + t) * 64 + lane]      = (bf16)nk0;
  Kb[((size_t)h * SEQ + t) * 64 + 32 + lane] = (bf16)nk1;
}

// ---------------------------------------------------------------------------
// Flash attention: one wave per (head, 16-query tile).
// S^T = K * Q^T  (C frag: lane=query, vgpr=key) ->  P^T B-frag built in-lane.
// y^T = V^T * P^T accumulated online; c_proj_scale folded into bf16 output.
// ---------------------------------------------------------------------------
__global__ __launch_bounds__(32)
void attn_kernel(const bf16* __restrict__ Qb, const bf16* __restrict__ Kb,
                 const bf16* __restrict__ Vt, const float* __restrict__ cps,
                 bf16* __restrict__ Yb) {
  const int h = blockIdx.x & 15;
  const int qbase = (blockIdx.x >> 4) * 16;
  const int lane = threadIdx.x;
  const int r = lane & 15, kg = lane >> 4;
  const int q = qbase + r;                       // this lane's query row

  const bf16* Qh = Qb + (size_t)h * SEQ * 64;
  const bf16* Kh = Kb + (size_t)h * SEQ * 64;
  const bf16* Vh = Vt + (size_t)h * 64 * SEQ;

  const v16bf qf0 = load_frag(Qh + (size_t)qbase * 64, 64);       // hd 0..31
  const v16bf qf1 = load_frag(Qh + (size_t)qbase * 64 + 32, 64);  // hd 32..63

  v8f yac[4];
#pragma unroll
  for (int i = 0; i < 4; i++) yac[i] = (v8f){0.f,0.f,0.f,0.f,0.f,0.f,0.f,0.f};
  float mrun = -3.0e38f, lrun = 0.0f;

  const int nkb = (qbase >> 5) + 1;              // 32-key blocks
  for (int kb = 0; kb < nkb; kb++) {
    const int kbb = kb * 32;

    if (kb + 1 < nkb) {                          // prefetch next K/V tiles
      __builtin_prefetch(Kh + (size_t)(kbb + 32) * 64 + r * 64, 0, 1);
      __builtin_prefetch(Vh + (size_t)(r * 4) * SEQ + kbb + 32, 0, 1);
    }

    v8f st0 = (v8f){0.f,0.f,0.f,0.f,0.f,0.f,0.f,0.f};
    v8f st1 = st0;
    st0 = wmma_bf16(load_frag(Kh + (size_t)kbb * 64, 64),        qf0, st0);
    st0 = wmma_bf16(load_frag(Kh + (size_t)kbb * 64 + 32, 64),   qf1, st0);
    st1 = wmma_bf16(load_frag(Kh + (size_t)(kbb+16) * 64, 64),      qf0, st1);
    st1 = wmma_bf16(load_frag(Kh + (size_t)(kbb+16) * 64 + 32, 64), qf1, st1);

    float s[16];
#pragma unroll
    for (int i = 0; i < 8; i++) {
      int key0 = kbb + 8 * kg + i;
      int key1 = key0 + 16;
      s[i]     = (key0 <= q) ? st0[i] * 0.125f : -3.0e38f;
      s[8 + i] = (key1 <= q) ? st1[i] * 0.125f : -3.0e38f;
    }
    float mloc = s[0];
#pragma unroll
    for (int i = 1; i < 16; i++) mloc = fmaxf(mloc, s[i]);
    mloc = fmaxf(mloc, __shfl_xor(mloc, 16, 32));  // lanes q / q+16 share a row
    float mnew = fmaxf(mrun, mloc);
    float alpha = __expf(mrun - mnew);
    float ls = 0.0f;
#pragma unroll
    for (int i = 0; i < 16; i++) { s[i] = __expf(s[i] - mnew); ls += s[i]; }
    ls += __shfl_xor(ls, 16, 32);
    lrun = lrun * alpha + ls;
    mrun = mnew;
#pragma unroll
    for (int i = 0; i < 8; i++) {
      yac[0][i] *= alpha; yac[1][i] *= alpha;
      yac[2][i] *= alpha; yac[3][i] *= alpha;
    }

    // P^T B-fragment (32 keys x 16 queries): purely in-lane repack
    union { v16bf v; unsigned int w[8]; } pf;
#pragma unroll
    for (int j = 0; j < 4; j++) {
      pf.w[j]     = pack2(s[2 * j],     s[2 * j + 1]);      // keys kg*8+2j(+1)
      pf.w[4 + j] = pack2(s[8 + 2 * j], s[8 + 2 * j + 1]);  // +16
    }

    // y^T += V^T(chunk) * P^T  for 4 hd-chunks of 16
#pragma unroll
    for (int c = 0; c < 4; c++)
      yac[c] = wmma_bf16(load_frag(Vh + (size_t)(c * 16) * SEQ + kbb, SEQ),
                         pf.v, yac[c]);
  }

  const float invl = 1.0f / lrun;
#pragma unroll
  for (int c = 0; c < 4; c++) {
#pragma unroll
    for (int rr = 0; rr < 8; rr += 2) {
      int col = h * 64 + c * 16 + rr + 8 * kg;
      float v0 = yac[c][rr]     * invl * cps[col];
      float v1 = yac[c][rr + 1] * invl * cps[col + 1];
      *(unsigned int*)(&Yb[(size_t)q * DMODEL + col]) = pack2(v0, v1);
    }
  }
}

// ---------------------------------------------------------------------------
// Launch
// ---------------------------------------------------------------------------
extern "C" void kernel_launch(void* const* d_in, const int* in_sizes, int n_in,
                              void* d_out, int out_size, void* d_ws, size_t ws_size,
                              hipStream_t stream) {
  const float* x      = (const float*)d_in[0];
  const float* ve     = (const float*)d_in[1];
  const float* c_q    = (const float*)d_in[2];
  const float* c_k    = (const float*)d_in[3];
  const float* c_v    = (const float*)d_in[4];
  const float* qkvs   = (const float*)d_in[5];
  const float* q_sc   = (const float*)d_in[6];
  const float* k_sc   = (const float*)d_in[7];
  const float* v_lam  = (const float*)d_in[8];
  const float* c_proj = (const float*)d_in[9];
  const float* cps    = (const float*)d_in[10];

  char* ws = (char*)d_ws;
  bf16*  Wqkv = (bf16*)(ws);                 // 3072*1024 bf16  = 6 MB
  bf16*  Xb   = (bf16*)(ws + 6291456);       // 2048*1024 bf16  = 4 MB
  bf16*  Wp   = (bf16*)(ws + 10485760);      // 1024*1024 bf16  = 2 MB
  float* cosT = (float*)(ws + 12582912);     // 2048*32 f32
  float* sinT = (float*)(ws + 12845056);
  float* qkvf = (float*)(ws + 13107200);     // 2048*3072 f32   = 24 MB
  bf16*  Qb   = (bf16*)(ws + 38273024);      // [H][T][64] bf16
  bf16*  Kb   = (bf16*)(ws + 42467328);
  bf16*  Vt   = (bf16*)(ws + 46661632);      // [H][64][T] bf16
  bf16*  Yb   = (bf16*)(ws + 50855936);      // [T][DMODEL] bf16 (scaled)

  prep_w_kernel<<<(3072 * 1024) / 256, 256, 0, stream>>>(c_q, c_k, c_v, qkvs, Wqkv);
  cvt_kernel<<<(2048 * 1024) / 256, 256, 0, stream>>>(x, Xb);
  cvt_kernel<<<(1024 * 1024) / 256, 256, 0, stream>>>(c_proj, Wp);
  rot_kernel<<<(2048 * 32) / 256, 256, 0, stream>>>(cosT, sinT);

  gemm_bf16_kernel<<<dim3(3072 / GBN, 2048 / GBM), 256, 2 * PAIR_B, stream>>>(
      Xb, Wqkv, qkvf, 2048, 3072, 1024);

  post_kernel<<<SEQ * NH, 32, 0, stream>>>(qkvf, ve, q_sc, k_sc, v_lam,
                                           cosT, sinT, Qb, Kb, Vt);

  attn_kernel<<<NH * (SEQ / 16), 32, 0, stream>>>(Qb, Kb, Vt, cps, Yb);

  gemm_bf16_kernel<<<dim3(1024 / GBN, 2048 / GBM), 256, 2 * PAIR_B, stream>>>(
      Yb, Wp, (float*)d_out, 2048, 1024, 1024);
}
